// Ptr_Net_26345329394336
// MI455X (gfx1250) — matrix operand order repaired and checked by reading, not compiled, not run on hardware
//
#include <hip/hip_runtime.h>
#include <hip/hip_bf16.h>
#include <cstdint>

#define BB   512
#define CITYN 100
#define HH   256
#define EE   256
#define DD2  512
#define G4   1024     // 4*H
#define PENF 1e6f

typedef __attribute__((ext_vector_type(16))) __bf16 v16bf;
typedef __attribute__((ext_vector_type(8)))  __bf16 v8bf;
typedef __attribute__((ext_vector_type(8)))  float  v8f;
typedef unsigned int u32;

__device__ __forceinline__ float sigmf(float x){ return 1.f/(1.f+expf(-x)); }

// -------------------- embedding: emb[b,c,e] = in[b,c,:2] . W_embed[e,:2] (bf16 out)
__global__ void k_embed(const float* __restrict__ in, const float* __restrict__ W,
                        __bf16* __restrict__ emb){
  long i = (long)blockIdx.x*blockDim.x + threadIdx.x;     // B*CITY*E threads
  long bc = i / EE; int e = (int)(i % EE);
  float x0 = in[bc*2+0], x1 = in[bc*2+1];
  emb[i] = (__bf16)(x0*W[e*2+0] + x1*W[e*2+1]);
}

// -------------------- pack [Wih;Whh]^T (or W^T when K2==0) fp32 -> bf16
// Output is in WMMA-B fragment-tiled layout: for each (ntile, ktile) 32x16 tile,
// lane = ((k>>4)&1)*16 + (n&15) owns 16 contiguous bf16 (i = k&15) => 32B/lane,
// loadable with two global_load_b128 directly into the B operand registers.
__global__ void k_pack(const float* __restrict__ Wih, const float* __restrict__ Whh,
                       int K1, int K2, int N, __bf16* __restrict__ out){
  long i = (long)blockIdx.x*blockDim.x + threadIdx.x;
  int Ktot = K1 + K2;
  long total = (long)Ktot*N;
  if (i >= total) return;
  int k = (int)(i / N), n = (int)(i % N);
  float w = (k < K1) ? Wih[(long)n*K1 + k] : Whh[(long)n*K2 + (k-K1)];
  int ktile = k >> 5, kh = (k >> 4) & 1, ii = k & 15;
  int ntile = n >> 4, col = n & 15;
  int lane = kh*16 + col;
  size_t idx = ((size_t)ntile * (Ktot >> 5) + ktile) * 512 + (size_t)lane*16 + ii;
  out[idx] = (__bf16)w;
}

// -------------------- bf16 WMMA GEMM: C[M,N] = [A1|A2] @ Bp + bias
// A row-major bf16 (two K-sources for the LSTM [x|h] concat), staged in LDS,
// fragment halves loaded as ds_load_b128. Bp is fragment-tiled (see k_pack),
// loaded directly from global (L2-resident weights). C fp32 row-major.
// Block tile 64x64, 4 waves, each wave 2x2 v_wmma_f32_16x16x32_bf16 tiles.
__global__ __launch_bounds__(128)
void k_gemm(const __bf16* __restrict__ A1, long lda1,
            const __bf16* __restrict__ A2, long lda2, int K1, int Ktot,
            const __bf16* __restrict__ Bp, int N,
            const float* __restrict__ bias, float* __restrict__ C, long ldc){
  __shared__ __align__(16) __bf16 As[64][40];   // 80B rows: 16B-aligned halves
  int tid = threadIdx.x, lane = tid & 31, wave = tid >> 5;
  int wm = wave >> 1, wn = wave & 1;
  int m0 = blockIdx.y * 64, n0 = blockIdx.x * 64;
  const v16bf* Bt = (const v16bf*)Bp;
  int ktiles = Ktot >> 5;
  v8f acc[2][2] = {};

  for (int kt = 0; kt < Ktot; kt += 32){
    const __bf16* Asrc; long lda; int kl;
    if (kt < K1){ Asrc = A1; lda = lda1; kl = kt; }
    else        { Asrc = A2; lda = lda2; kl = kt - K1; }
    // stage A 64x32 (dword loads: 2 bf16 each)
    for (int i = tid; i < 64*16; i += 128){
      int rr = i >> 4, cc = i & 15;
      u32 u = *(const u32*)(Asrc + (long)(m0+rr)*lda + kl + cc*2);
      *(u32*)(&As[rr][cc*2]) = u;
    }
    // B fragments: two b128 global loads per tile, straight from packed layout
    v16bf bfr[2];
    int kti = kt >> 5;
    #pragma unroll
    for (int nt = 0; nt < 2; ++nt){
      int ntile = (n0 >> 4) + wn*2 + nt;
      bfr[nt] = Bt[((size_t)ntile * ktiles + kti) * 32 + lane];
    }
    __syncthreads();

    // A fragments: lane = row (lane&15), K-half select (lane>>4);
    // VGPR0-3 hold K=kb..kb+7, VGPR4-7 hold K=kb+8..kb+15 -> two ds_load_b128
    v16bf af[2];
    int kbA = (lane >> 4) * 16;
    #pragma unroll
    for (int mt = 0; mt < 2; ++mt){
      int r = wm*32 + mt*16 + (lane & 15);
      v8bf alo = *(const v8bf*)&As[r][kbA];
      v8bf ahi = *(const v8bf*)&As[r][kbA + 8];
      af[mt] = __builtin_shufflevector(alo, ahi, 0,1,2,3,4,5,6,7,8,9,10,11,12,13,14,15);
    }
    #pragma unroll
    for (int mt = 0; mt < 2; ++mt)
      #pragma unroll
      for (int nt = 0; nt < 2; ++nt)
        acc[mt][nt] = __builtin_amdgcn_wmma_f32_16x16x32_bf16(
            false, af[mt], false, bfr[nt], (short)0, acc[mt][nt], false, false);
    __syncthreads();
  }

  // bias: one load per nt-tile per lane (bias is never null; dummy zeros otherwise)
  float badd[2];
  #pragma unroll
  for (int nt = 0; nt < 2; ++nt)
    badd[nt] = bias[n0 + wn*32 + nt*16 + (lane & 15)];

  // C layout: VGPR v -> M = v + 8*(lane/16), N = lane%16
  #pragma unroll
  for (int mt = 0; mt < 2; ++mt)
    #pragma unroll
    for (int nt = 0; nt < 2; ++nt)
      #pragma unroll
      for (int v = 0; v < 8; ++v){
        int row = m0 + wm*32 + mt*16 + v + 8*(lane >> 4);
        int col = n0 + wn*32 + nt*16 + (lane & 15);
        C[(long)row*ldc + col] = acc[mt][nt][v] + badd[nt];
      }
}

// -------------------- LSTM pointwise: gates[B,4H] -> update c, h; scatter h
__global__ void k_lstm_update(const float* __restrict__ gates, float* __restrict__ c,
                              __bf16* __restrict__ hstate,
                              float* __restrict__ outf, __bf16* __restrict__ outb,
                              long out_off, long out_ld){
  long i = (long)blockIdx.x*blockDim.x + threadIdx.x;    // B*H threads
  long b = i / HH; int n = (int)(i % HH);
  const float* g = gates + b*G4;
  float gi = g[n], gf = g[HH+n], gg = g[2*HH+n], go = g[3*HH+n];
  float cn = sigmf(gf)*c[i] + sigmf(gi)*tanhf(gg);
  float h  = sigmf(go)*tanhf(cn);
  c[i] = cn;
  hstate[i] = (__bf16)h;
  long oi = b*out_ld + out_off + n;
  if (outf) outf[oi] = h;
  if (outb) outb[oi] = (__bf16)h;
}

// -------------------- fused decoder attention / softmax / argmax / CE / context
__global__ __launch_bounds__(256)
void k_attention(const float* __restrict__ ref1, const float* __restrict__ ref2,
                 const float* __restrict__ q, const float* __restrict__ q2,
                 const float* __restrict__ v1, const float* __restrict__ v2,
                 const float* __restrict__ Enc, float* __restrict__ played,
                 float* __restrict__ loss, const int* __restrict__ roads, int t,
                 __bf16* __restrict__ xout){
  __shared__ float sq[DD2], sq2[DD2];
  __shared__ float sow[CITYN], saw[CITYN];
  __shared__ float sred[2];
  int b = blockIdx.x, tid = threadIdx.x, lane = tid & 31, wave = tid >> 5;

  sq [tid]     = q [(long)b*DD2 + tid];
  sq [tid+256] = q [(long)b*DD2 + tid + 256];
  sq2[tid]     = q2[(long)b*DD2 + tid];
  sq2[tid+256] = q2[(long)b*DD2 + tid + 256];
  __syncthreads();

  // ow/aw scores: wave per city, lane-parallel over D2, wave32 butterfly reduce
  for (int cc = wave; cc < CITYN; cc += 8){
    const float* r1 = ref1 + ((long)b*CITYN + cc)*DD2;
    const float* r2 = ref2 + ((long)b*CITYN + cc)*DD2;
    float s1 = 0.f, s2 = 0.f;
    for (int n = lane; n < DD2; n += 32){
      s1 += tanhf(r1[n] + sq[n])  * v1[n];
      s2 += tanhf(r2[n] + sq2[n]) * v2[n];
    }
    for (int o = 16; o > 0; o >>= 1){
      s1 += __shfl_xor(s1, o, 32);
      s2 += __shfl_xor(s2, o, 32);
    }
    if (lane == 0){
      sow[cc] = s1 - PENF * played[(long)b*CITYN + cc];
      saw[cc] = s2;
    }
  }
  __syncthreads();

  if (wave == 0){
    // argmax + logsumexp over ow; CE and played update
    float bv = -3.4e38f; int bi = 0;
    for (int cc = lane; cc < CITYN; cc += 32){ float x = sow[cc]; if (x > bv){ bv = x; bi = cc; } }
    for (int o = 16; o > 0; o >>= 1){
      float ov = __shfl_xor(bv, o, 32); int oi = __shfl_xor(bi, o, 32);
      if (ov > bv || (ov == bv && oi < bi)){ bv = ov; bi = oi; }
    }
    float s = 0.f;
    for (int cc = lane; cc < CITYN; cc += 32) s += expf(sow[cc] - bv);
    for (int o = 16; o > 0; o >>= 1) s += __shfl_xor(s, o, 32);
    if (lane == 0){
      float lse = bv + logf(s);
      int tgt = roads[(long)b*CITYN + t];
      loss[b] += lse - sow[tgt];
      played[(long)b*CITYN + bi] += 1.f;
    }
  } else if (wave == 1){
    // softmax stats for aw
    float mv = -3.4e38f;
    for (int cc = lane; cc < CITYN; cc += 32) mv = fmaxf(mv, saw[cc]);
    for (int o = 16; o > 0; o >>= 1) mv = fmaxf(mv, __shfl_xor(mv, o, 32));
    float s = 0.f;
    for (int cc = lane; cc < CITYN; cc += 32) s += expf(saw[cc] - mv);
    for (int o = 16; o > 0; o >>= 1) s += __shfl_xor(s, o, 32);
    if (lane == 0){ sred[0] = mv; sred[1] = 1.f / s; }
  }
  __syncthreads();
  if (tid < CITYN) saw[tid] = expf(saw[tid] - sred[0]) * sred[1];
  __syncthreads();

  // context: x_new = aw @ Enc[b]  (written as bf16 for next step's GEMMs)
  for (int n = tid; n < DD2; n += 256){
    float acc = 0.f;
    const float* e = Enc + (long)b*CITYN*DD2 + n;
    #pragma unroll 4
    for (int cc = 0; cc < CITYN; ++cc) acc += saw[cc] * e[(long)cc*DD2];
    xout[(long)b*DD2 + n] = (__bf16)acc;
  }
}

// ==================== host orchestration ====================
extern "C" void kernel_launch(void* const* d_in, const int* in_sizes, int n_in,
                              void* d_out, int out_size, void* d_ws, size_t ws_size,
                              hipStream_t stream){
  (void)in_sizes; (void)n_in; (void)out_size; (void)ws_size;
  const float* inputs  = (const float*)d_in[0];
  const int*   roads   = (const int*)  d_in[1];
  const float* W_embed = (const float*)d_in[2];
  const float* eWihf=(const float*)d_in[3],  *eWhhf=(const float*)d_in[4],  *ebf=(const float*)d_in[5];
  const float* eWihb=(const float*)d_in[6],  *eWhhb=(const float*)d_in[7],  *ebb=(const float*)d_in[8];
  const float* dWihf=(const float*)d_in[9],  *dWhhf=(const float*)d_in[10], *dbf=(const float*)d_in[11];
  const float* dWihb=(const float*)d_in[12], *dWhhb=(const float*)d_in[13], *dbb=(const float*)d_in[14];
  const float* Wref =(const float*)d_in[15], *Wq  =(const float*)d_in[16], *v1=(const float*)d_in[17];
  const float* Wref2=(const float*)d_in[18], *Wq2 =(const float*)d_in[19], *v2=(const float*)d_in[20];
  float* loss = (float*)d_out;

  // bump allocator over d_ws (256B aligned)
  char* base = (char*)d_ws; size_t off = 0;
  auto alloc = [&](size_t bytes)->void*{
    void* p = base + off; off = (off + bytes + 255) & ~(size_t)255; return p; };

  __bf16* emb    = (__bf16*)alloc((size_t)BB*CITYN*EE*2);
  __bf16* WcEf   = (__bf16*)alloc((size_t)(EE+HH)*G4*2);
  __bf16* WcEb   = (__bf16*)alloc((size_t)(EE+HH)*G4*2);
  __bf16* WcDf   = (__bf16*)alloc((size_t)(DD2+HH)*G4*2);
  __bf16* WcDb   = (__bf16*)alloc((size_t)(DD2+HH)*G4*2);
  __bf16* WrefT  = (__bf16*)alloc((size_t)DD2*DD2*2);
  __bf16* Wref2T = (__bf16*)alloc((size_t)DD2*DD2*2);
  __bf16* WqT    = (__bf16*)alloc((size_t)DD2*DD2*2);
  __bf16* Wq2T   = (__bf16*)alloc((size_t)DD2*DD2*2);
  float*  Enc    = (float*) alloc((size_t)BB*CITYN*DD2*4);
  __bf16* Encb   = (__bf16*)alloc((size_t)BB*CITYN*DD2*2);
  float*  ref1   = (float*) alloc((size_t)BB*CITYN*DD2*4);
  float*  ref2   = (float*) alloc((size_t)BB*CITYN*DD2*4);
  float*  gatesF = (float*) alloc((size_t)BB*G4*4);
  float*  gatesB = (float*) alloc((size_t)BB*G4*4);
  __bf16* hf     = (__bf16*)alloc((size_t)BB*HH*2);
  __bf16* hb     = (__bf16*)alloc((size_t)BB*HH*2);
  float*  cf     = (float*) alloc((size_t)BB*HH*4);
  float*  cb     = (float*) alloc((size_t)BB*HH*4);
  __bf16* decout = (__bf16*)alloc((size_t)BB*DD2*2);
  __bf16* xb16   = (__bf16*)alloc((size_t)BB*DD2*2);
  float*  qbuf   = (float*) alloc((size_t)BB*DD2*4);
  float*  qbuf2  = (float*) alloc((size_t)BB*DD2*4);
  float*  played = (float*) alloc((size_t)BB*CITYN*4);
  float*  zbias  = (float*) alloc((size_t)G4*4);

  // zero-init state (graph-capturable)
  hipMemsetAsync(xb16,   0, (size_t)BB*DD2*2,   stream);
  hipMemsetAsync(hf,     0, (size_t)BB*HH*2,    stream);
  hipMemsetAsync(hb,     0, (size_t)BB*HH*2,    stream);
  hipMemsetAsync(cf,     0, (size_t)BB*HH*4,    stream);
  hipMemsetAsync(cb,     0, (size_t)BB*HH*4,    stream);
  hipMemsetAsync(played, 0, (size_t)BB*CITYN*4, stream);
  hipMemsetAsync(zbias,  0, (size_t)G4*4,       stream);
  hipMemsetAsync(loss,   0, (size_t)BB*4,       stream);

  // pack weights to bf16 fragment-tiled B layout
  k_pack<<<((EE+HH)*G4)/256, 256, 0, stream>>>(eWihf, eWhhf, EE,  HH, G4,  WcEf);
  k_pack<<<((EE+HH)*G4)/256, 256, 0, stream>>>(eWihb, eWhhb, EE,  HH, G4,  WcEb);
  k_pack<<<((DD2+HH)*G4)/256,256, 0, stream>>>(dWihf, dWhhf, DD2, HH, G4,  WcDf);
  k_pack<<<((DD2+HH)*G4)/256,256, 0, stream>>>(dWihb, dWhhb, DD2, HH, G4,  WcDb);
  k_pack<<<(DD2*DD2)/256,    256, 0, stream>>>(Wref,  nullptr, DD2, 0, DD2, WrefT);
  k_pack<<<(DD2*DD2)/256,    256, 0, stream>>>(Wref2, nullptr, DD2, 0, DD2, Wref2T);
  k_pack<<<(DD2*DD2)/256,    256, 0, stream>>>(Wq,    nullptr, DD2, 0, DD2, WqT);
  k_pack<<<(DD2*DD2)/256,    256, 0, stream>>>(Wq2,   nullptr, DD2, 0, DD2, Wq2T);

  // embedding
  k_embed<<<((long)BB*CITYN*EE)/256, 256, 0, stream>>>(inputs, W_embed, emb);

  auto gemm = [&](float* C, long ldc, int M, int N,
                  const __bf16* A1, long lda1, int K1,
                  const __bf16* A2, long lda2, int K2,
                  const __bf16* Bm, const float* bias){
    dim3 g(N/64, M/64);
    k_gemm<<<g, 128, 0, stream>>>(A1, lda1, A2, lda2, K1, K1+K2, Bm, N, bias, C, ldc);
  };

  // ---------------- encoder (forward then backward) ----------------
  for (int t = 0; t < CITYN; ++t){
    gemm(gatesF, G4, BB, G4, emb + (long)t*EE, (long)CITYN*EE, EE, hf, HH, HH, WcEf, ebf);
    k_lstm_update<<<(BB*HH)/256, 256, 0, stream>>>(gatesF, cf, hf, Enc, Encb,
                                                   (long)t*DD2, (long)CITYN*DD2);
  }
  for (int t = CITYN-1; t >= 0; --t){
    gemm(gatesB, G4, BB, G4, emb + (long)t*EE, (long)CITYN*EE, EE, hb, HH, HH, WcEb, ebb);
    k_lstm_update<<<(BB*HH)/256, 256, 0, stream>>>(gatesB, cb, hb, Enc, Encb,
                                                   (long)t*DD2 + HH, (long)CITYN*DD2);
  }

  // ---------------- hoisted reference projections (51200x512x512) ----------------
  gemm(ref1, DD2, BB*CITYN, DD2, Encb, DD2, DD2, nullptr, 0, 0, WrefT,  zbias);
  gemm(ref2, DD2, BB*CITYN, DD2, Encb, DD2, DD2, nullptr, 0, 0, Wref2T, zbias);

  // ---------------- decoder ----------------
  // h/c state buffers carry over from encoder finals (matches jax scan carries)
  for (int t = 0; t < CITYN; ++t){
    gemm(gatesF, G4, BB, G4, xb16, DD2, DD2, hf, HH, HH, WcDf, dbf);
    gemm(gatesB, G4, BB, G4, xb16, DD2, DD2, hb, HH, HH, WcDb, dbb);
    gemm(qbuf2,  DD2, BB, DD2, xb16, DD2, DD2, nullptr, 0, 0, Wq2T, zbias); // q2 from pre-update x
    k_lstm_update<<<(BB*HH)/256, 256, 0, stream>>>(gatesF, cf, hf, nullptr, decout, 0,  DD2);
    k_lstm_update<<<(BB*HH)/256, 256, 0, stream>>>(gatesB, cb, hb, nullptr, decout, HH, DD2);
    gemm(qbuf, DD2, BB, DD2, decout, DD2, DD2, nullptr, 0, 0, WqT, zbias);
    k_attention<<<BB, 256, 0, stream>>>(ref1, ref2, qbuf, qbuf2, v1, v2, Enc,
                                        played, loss, roads, t, xb16);
  }
}
